// WTAGNNLayer_81716047774294
// MI455X (gfx1250) — matrix operand order repaired and checked by reading, not compiled, not run on hardware
//
#include <hip/hip_runtime.h>
#include <hip/hip_bf16.h>

typedef __attribute__((ext_vector_type(16))) _Float16 v16h;
typedef __attribute__((ext_vector_type(8)))  _Float16 v8h;
typedef __attribute__((ext_vector_type(4)))  _Float16 v4h;
typedef __attribute__((ext_vector_type(8)))  float    v8f;
typedef __attribute__((ext_vector_type(4)))  float    v4f;

#define D 128

// ---------------------------------------------------------------------------
// A-fragment loader: 16-bit A matrix 16x32 (MxK) per CDNA5 ISA layout.
// lane<16:  M=lane,    elems 0-7 -> K=kBase..+7,   elems 8-15 -> K=kBase+16..+23
// lane>=16: M=lane-16, elems 0-7 -> K=kBase+8..+15, elems 8-15 -> K=kBase+24..+31
// Caller passes krun = kBase + (lane>>4)*8; Arow = &A[row*D]. 16B-aligned runs.
// ---------------------------------------------------------------------------
__device__ __forceinline__ v16h wmma_a_from_global(const float* __restrict__ Arow,
                                                   int krun) {
  const float* p = Arow + krun;
  v4f x0 = *(const v4f*)(p);
  v4f x1 = *(const v4f*)(p + 4);
  v4f x2 = *(const v4f*)(p + 16);
  v4f x3 = *(const v4f*)(p + 20);
  v16h a;
#pragma unroll
  for (int e = 0; e < 4; ++e) {
    a[e]      = (_Float16)x0[e];
    a[e + 4]  = (_Float16)x1[e];
    a[e + 8]  = (_Float16)x2[e];
    a[e + 12] = (_Float16)x3[e];
  }
  return a;
}

__device__ __forceinline__ v16h wmma_b_frag(const _Float16* __restrict__ Wswz,
                                            int kt, int nt, int lane) {
  return *(const v16h*)(Wswz + ((size_t)((kt * 8 + nt) * 32 + lane)) * 16);
}

// ---------------------------------------------------------------------------
// Pre-swizzle a row-major f32 weight [K x 128] into f16 B-fragments.
// Fragment f = kt*8 + nt. lane L elem e = W[kt*32 + (L>>4)*8 + e + (e&8)][nt*16 + (L&15)]
// so the GEMM loads one contiguous 32B v16h per lane per (kt,nt).
// ---------------------------------------------------------------------------
__global__ void k_swz(const float* __restrict__ W, _Float16* __restrict__ out, int K) {
  int tid = blockIdx.x * blockDim.x + threadIdx.x;
  int nfrag = (K >> 5) * 8;
  if (tid >= nfrag * 32) return;
  int f = tid >> 5, L = tid & 31;
  int kt = f >> 3, nt = f & 7;
  int kb = kt * 32 + (L >> 4) * 8;
  int nb = nt * 16 + (L & 15);
  v16h val;
#pragma unroll
  for (int e = 0; e < 16; ++e)
    val[e] = (_Float16)W[(size_t)(kb + e + (e & 8)) * D + nb];
  *(v16h*)(out + (size_t)tid * 16) = val;
}

__global__ void k_zero(float* __restrict__ p, int n) {
  int i = blockIdx.x * blockDim.x + threadIdx.x;
  if (i < n) p[i] = 0.0f;
}

__global__ void k_deg(const int* __restrict__ dst, float* __restrict__ deg, int E) {
  int i = blockIdx.x * blockDim.x + threadIdx.x;
  if (i < E) atomicAdd(&deg[dst[i]], 1.0f);
}

// nb_ef = s / max(deg,1), in place on s
__global__ void k_mean(float* __restrict__ s, const float* __restrict__ deg, int n) {
  int i = blockIdx.x * blockDim.x + threadIdx.x;
  if (i < n) {
    float d = deg[i >> 7];
    s[i] = s[i] / (d > 1.0f ? d : 1.0f);
  }
}

// ---------------------------------------------------------------------------
// nf2 = relu(nf @ W_node + node_bias).  One wave = one 16x128 strip.
// A fragments preloaded once; nt-outer loop with per-tile epilogue.  Full
// strips take a guard-free store path (base ptr + immediate offsets).
// ---------------------------------------------------------------------------
__global__ void __launch_bounds__(256) k_node_gemm(const float* __restrict__ nf,
                                                   const _Float16* __restrict__ Wswz,
                                                   const float* __restrict__ bias,
                                                   float* __restrict__ nf2, int M) {
  const int lane  = threadIdx.x & 31;
  const int wave  = threadIdx.x >> 5;
  const int m     = (blockIdx.x * 8 + wave) * 16;
  const int half8 = (lane >> 4) * 8;
  const bool full = (m + 16 <= M);
  int row = m + (lane & 15);
  if (row >= M) row = M - 1;
  const float* Arow = nf + (size_t)row * D;

  v16h a0 = wmma_a_from_global(Arow, 0 * 32 + half8);
  v16h a1 = wmma_a_from_global(Arow, 1 * 32 + half8);
  v16h a2 = wmma_a_from_global(Arow, 2 * 32 + half8);
  v16h a3 = wmma_a_from_global(Arow, 3 * 32 + half8);

  const int lc    = lane & 15;
  const int rbase = m + (lane >> 4) * 8;          // C layout: vgpr r -> M=r (+8 hi)
#pragma unroll
  for (int nt = 0; nt < 8; ++nt) {
    v8f c = {};
    c = __builtin_amdgcn_wmma_f32_16x16x32_f16(false, a0, false, wmma_b_frag(Wswz, 0, nt, lane), (short)0, c, false, false);
    c = __builtin_amdgcn_wmma_f32_16x16x32_f16(false, a1, false, wmma_b_frag(Wswz, 1, nt, lane), (short)0, c, false, false);
    c = __builtin_amdgcn_wmma_f32_16x16x32_f16(false, a2, false, wmma_b_frag(Wswz, 2, nt, lane), (short)0, c, false, false);
    c = __builtin_amdgcn_wmma_f32_16x16x32_f16(false, a3, false, wmma_b_frag(Wswz, 3, nt, lane), (short)0, c, false, false);
    const int col  = nt * 16 + lc;
    const float bc = bias[col];
    float* outp = nf2 + (size_t)rbase * D + col;
    if (full) {                                   // guard-free: imm-offset stores
#pragma unroll
      for (int r = 0; r < 8; ++r) {
        float v = c[r] + bc;
        outp[r * D] = v > 0.0f ? v : 0.0f;
      }
    } else {
#pragma unroll
      for (int r = 0; r < 8; ++r) {
        if (rbase + r < M) {
          float v = c[r] + bc;
          outp[r * D] = v > 0.0f ? v : 0.0f;
        }
      }
    }
  }
}

// ---------------------------------------------------------------------------
// Pass 1 over edges: ef1 strip = ef @ W_edge (never hits HBM), scatter-add
// into s[dst] (L2-resident).  Same structure; full strips skip per-row guards.
// ---------------------------------------------------------------------------
__global__ void __launch_bounds__(256) k_edge_scatter(const float* __restrict__ ef,
                                                      const _Float16* __restrict__ Wswz,
                                                      const int* __restrict__ dst,
                                                      float* __restrict__ s, int E) {
  const int lane  = threadIdx.x & 31;
  const int wave  = threadIdx.x >> 5;
  const int m     = (blockIdx.x * 8 + wave) * 16;
  const int half8 = (lane >> 4) * 8;
  const bool full = (m + 16 <= E);
  if (m + 16 < E) __builtin_prefetch(ef + (size_t)(m + 16) * D, 0, 0);
  int row = m + (lane & 15);
  if (row >= E) row = E - 1;
  const float* Arow = ef + (size_t)row * D;

  v16h a0 = wmma_a_from_global(Arow, 0 * 32 + half8);
  v16h a1 = wmma_a_from_global(Arow, 1 * 32 + half8);
  v16h a2 = wmma_a_from_global(Arow, 2 * 32 + half8);
  v16h a3 = wmma_a_from_global(Arow, 3 * 32 + half8);

  const int lc    = lane & 15;
  const int rbase = m + (lane >> 4) * 8;
  size_t nrow[8];                                  // hoisted dst gathers
#pragma unroll
  for (int r = 0; r < 8; ++r) {
    int e = rbase + r;
    nrow[r] = (size_t)dst[e < E ? e : (E - 1)] * D;
  }

#pragma unroll
  for (int nt = 0; nt < 8; ++nt) {
    v8f c = {};
    c = __builtin_amdgcn_wmma_f32_16x16x32_f16(false, a0, false, wmma_b_frag(Wswz, 0, nt, lane), (short)0, c, false, false);
    c = __builtin_amdgcn_wmma_f32_16x16x32_f16(false, a1, false, wmma_b_frag(Wswz, 1, nt, lane), (short)0, c, false, false);
    c = __builtin_amdgcn_wmma_f32_16x16x32_f16(false, a2, false, wmma_b_frag(Wswz, 2, nt, lane), (short)0, c, false, false);
    c = __builtin_amdgcn_wmma_f32_16x16x32_f16(false, a3, false, wmma_b_frag(Wswz, 3, nt, lane), (short)0, c, false, false);
    const int col = nt * 16 + lc;
    if (full) {
#pragma unroll
      for (int r = 0; r < 8; ++r)
        atomicAdd(&s[nrow[r] + col], c[r]);
    } else {
#pragma unroll
      for (int r = 0; r < 8; ++r)
        if (rbase + r < E) atomicAdd(&s[nrow[r] + col], c[r]);
    }
  }
}

// ---------------------------------------------------------------------------
// Pass 2 over edges (fully fused): recompute ef1 strip, add nb_ef[dst], build
// the 16x256 concatenated activation in wave-private LDS (f16, padded stride),
// then a K=256 WMMA chain against swizzled dense_W.  4 waves/block -> 33.8 KB.
// ---------------------------------------------------------------------------
__global__ void __launch_bounds__(128) k_edge_out(const float* __restrict__ ef,
                                                  const _Float16* __restrict__ We,
                                                  const _Float16* __restrict__ Wd,
                                                  const float* __restrict__ nb,
                                                  const float* __restrict__ nf2,
                                                  const int* __restrict__ src,
                                                  const int* __restrict__ dst,
                                                  const float* __restrict__ dense_b,
                                                  const float* __restrict__ edge_bias,
                                                  float* __restrict__ ef2, int E) {
  // per-wave: cols 0..127 tmp_ef, 128..255 srcdst; +8 pad halves (528B stride)
  __shared__ __align__(16) _Float16 As[4][16][264];

  const int lane  = threadIdx.x & 31;
  const int wave  = threadIdx.x >> 5;               // 0..3
  const int m     = (blockIdx.x * 4 + wave) * 16;
  const int half8 = (lane >> 4) * 8;
  const bool full = (m + 16 <= E);
  if (m + 16 < E) __builtin_prefetch(ef + (size_t)(m + 16) * D, 0, 0);
  int row = m + (lane & 15);
  if (row >= E) row = E - 1;
  const float* Arow = ef + (size_t)row * D;
  _Float16 (*A)[264] = As[wave];
  const int lc = lane & 15;

  // ---- step 1: tmp_ef strip = (ef @ W_edge) + nb_ef[dst]  -> LDS (f16)
  {
    v16h a0 = wmma_a_from_global(Arow, 0 * 32 + half8);
    v16h a1 = wmma_a_from_global(Arow, 1 * 32 + half8);
    v16h a2 = wmma_a_from_global(Arow, 2 * 32 + half8);
    v16h a3 = wmma_a_from_global(Arow, 3 * 32 + half8);

    size_t nrow[8];
#pragma unroll
    for (int r = 0; r < 8; ++r) {
      int e = m + (lane >> 4) * 8 + r;
      if (e >= E) e = E - 1;
      nrow[r] = (size_t)dst[e] * D;
    }

#pragma unroll
    for (int nt = 0; nt < 8; ++nt) {
      v8f c = {};
      c = __builtin_amdgcn_wmma_f32_16x16x32_f16(false, a0, false, wmma_b_frag(We, 0, nt, lane), (short)0, c, false, false);
      c = __builtin_amdgcn_wmma_f32_16x16x32_f16(false, a1, false, wmma_b_frag(We, 1, nt, lane), (short)0, c, false, false);
      c = __builtin_amdgcn_wmma_f32_16x16x32_f16(false, a2, false, wmma_b_frag(We, 2, nt, lane), (short)0, c, false, false);
      c = __builtin_amdgcn_wmma_f32_16x16x32_f16(false, a3, false, wmma_b_frag(We, 3, nt, lane), (short)0, c, false, false);
      const int col = nt * 16 + lc;
#pragma unroll
      for (int r = 0; r < 8; ++r) {
        int rr = (lane >> 4) * 8 + r;               // local row 0..15
        A[rr][col] = (_Float16)(c[r] + nb[nrow[r] + col]);
      }
    }
  }

  // ---- step 2: srcdst = 0.5*(nf2[src] + nf2[dst]) -> LDS cols 128..255
  {
    int rr = lane >> 1;                             // 2 lanes per row
    int c0 = (lane & 1) * 64;                       // 64 cols each
    int e  = m + rr;
    if (e >= E) e = E - 1;
    const float* ps = nf2 + (size_t)src[e] * D + c0;
    const float* pd = nf2 + (size_t)dst[e] * D + c0;
#pragma unroll
    for (int j = 0; j < 16; ++j) {
      v4f a4 = *(const v4f*)(ps + j * 4);
      v4f b4 = *(const v4f*)(pd + j * 4);
      v4h h;
#pragma unroll
      for (int i = 0; i < 4; ++i) h[i] = (_Float16)(0.5f * (a4[i] + b4[i]));
      *(v4h*)&A[rr][128 + c0 + j * 4] = h;
    }
  }
  __syncthreads();   // order cross-lane LDS writes before fragment reads

  // ---- step 3: ef2 strip = relu(As(16x256) @ dense_W + dense_b + edge_bias)
  {
    const int arow = lane & 15;
    v16h a[8];
#pragma unroll
    for (int kt = 0; kt < 8; ++kt) {                // preload all A frags from LDS
      const _Float16* pr = &A[arow][kt * 32 + half8];  // 16B-aligned runs
      v8h lo = *(const v8h*)pr;
      v8h hi = *(const v8h*)(pr + 16);
#pragma unroll
      for (int e2 = 0; e2 < 8; ++e2) { a[kt][e2] = lo[e2]; a[kt][8 + e2] = hi[e2]; }
    }
    const int rbase = m + (lane >> 4) * 8;
#pragma unroll
    for (int nt = 0; nt < 8; ++nt) {
      v8f c = {};
#pragma unroll
      for (int kt = 0; kt < 8; ++kt)
        c = __builtin_amdgcn_wmma_f32_16x16x32_f16(false, a[kt], false,
                                                   wmma_b_frag(Wd, kt, nt, lane),
                                                   (short)0, c, false, false);
      const int col  = nt * 16 + lc;
      const float bc = dense_b[col] + edge_bias[col];
      float* outp = ef2 + (size_t)rbase * D + col;
      if (full) {                                   // guard-free: imm-offset stores
#pragma unroll
        for (int r = 0; r < 8; ++r) {
          float v = c[r] + bc;
          outp[r * D] = v > 0.0f ? v : 0.0f;
        }
      } else {
#pragma unroll
        for (int r = 0; r < 8; ++r) {
          if (rbase + r < E) {
            float v = c[r] + bc;
            outp[r * D] = v > 0.0f ? v : 0.0f;
          }
        }
      }
    }
  }
}

// ---------------------------------------------------------------------------
extern "C" void kernel_launch(void* const* d_in, const int* in_sizes, int n_in,
                              void* d_out, int out_size, void* d_ws, size_t ws_size,
                              hipStream_t stream) {
  const float* nf        = (const float*)d_in[0];
  const float* ef        = (const float*)d_in[1];
  const float* W_node    = (const float*)d_in[2];
  const float* W_edge    = (const float*)d_in[3];
  const float* node_bias = (const float*)d_in[4];
  const float* edge_bias = (const float*)d_in[5];
  const float* dense_W   = (const float*)d_in[6];
  const float* dense_b   = (const float*)d_in[7];
  const int*   src       = (const int*)d_in[8];
  const int*   dst       = (const int*)d_in[9];

  const int Nn = in_sizes[0] / D;   // 50000
  const int E  = in_sizes[8];       // 800000

  float* nf2 = (float*)d_out;
  float* ef2 = nf2 + (size_t)Nn * D;

  // Workspace: s[Nn*D] | deg[Nn] | Wn_swz | We_swz | Wd_swz   (~26 MB total)
  float* s   = (float*)d_ws;
  float* deg = s + (size_t)Nn * D;
  size_t off = (size_t)Nn * D + (size_t)Nn;
  off = (off + 15) & ~(size_t)15;                 // 64B align
  _Float16* Wn = (_Float16*)(s + off);
  _Float16* We = Wn + 128 * D;
  _Float16* Wd = We + 128 * D;

  const int nz     = Nn * D + Nn;
  const int nStrip = (Nn + 15) / 16;              // 3125
  const int eStrip = (E + 15) / 16;               // 50000

  hipLaunchKernelGGL(k_zero, dim3((nz + 255) / 256), dim3(256), 0, stream, s, nz);
  hipLaunchKernelGGL(k_swz,  dim3(4), dim3(256), 0, stream, W_node,  Wn, 128);
  hipLaunchKernelGGL(k_swz,  dim3(4), dim3(256), 0, stream, W_edge,  We, 128);
  hipLaunchKernelGGL(k_swz,  dim3(8), dim3(256), 0, stream, dense_W, Wd, 256);
  hipLaunchKernelGGL(k_deg,  dim3((E + 255) / 256), dim3(256), 0, stream, dst, deg, E);

  hipLaunchKernelGGL(k_node_gemm, dim3((nStrip + 7) / 8), dim3(256), 0, stream,
                     nf, Wn, node_bias, nf2, Nn);
  hipLaunchKernelGGL(k_edge_scatter, dim3((eStrip + 7) / 8), dim3(256), 0, stream,
                     ef, We, dst, s, E);
  hipLaunchKernelGGL(k_mean, dim3((Nn * D + 255) / 256), dim3(256), 0, stream,
                     s, deg, Nn * D);
  hipLaunchKernelGGL(k_edge_out, dim3((eStrip + 3) / 4), dim3(128), 0, stream,
                     ef, We, Wd, s, nf2, src, dst, dense_b, edge_bias, ef2, E);
}
